// SparseLinearModule_26568667693718
// MI455X (gfx1250) — matrix operand order repaired and checked by reading, not compiled, not run on hardware
//
#include <hip/hip_runtime.h>

// ---------------------------------------------------------------------------
// SparseLinear: out = X @ (W + scatter(sparse))^T + bias
//   X: 16384x4096 f32, W: 4096x4096 f32, out: 16384x4096 f32
// Strategy:
//   1) W_eff = W + scatter(sparse) in d_ws (64 MB, L2 resident)
//   2) One-time split to bf16 hi/lo planes (Xh/Xl and Wh/Wl both in plain K
//      order) -> GEMM loop has zero conversion VALU.
//   3) bf16x3 split GEMM on V_WMMA_F32_16X16X32_BF16 (acc f32, ~2^-16 rel err).
//      Tiles move global->LDS via GLOBAL_LOAD_ASYNC_TO_LDS_B128 with a 2-deep
//      double-buffered pipeline: issue slab s+1, s_wait_asynccnt 8 retires
//      slab s (in-order completion), compute overlaps copy latency.
//   Fallback (small ws): on-the-fly split kernel (needs only 64 MB).
// ---------------------------------------------------------------------------

#define OUT_F 4096
#define IN_F  4096
#define M_TOK 16384

typedef __attribute__((ext_vector_type(8)))  float  v8f;
typedef __attribute__((ext_vector_type(4)))  __bf16 v4bf;
typedef __attribute__((ext_vector_type(16))) __bf16 v16bf;

#define BM 128
#define BN 128
#define BK 32
#define BKP 40   // padded LDS row stride (bf16): 80 B = 5x16 B, b128-aligned
#define NSLAB (IN_F / BK)

// 16-bit WMMA A-matrix (16x32) lane layout (ISA 7.12.2): half0 lanes hold
// K {0..7,16..23}, half1 {8..15,24..31}. Permute K within each 32-block to
// [0..7,16..23,8..15,24..31] so each half is one contiguous 32 B run.
__device__ __forceinline__ int permA(int k) {
  return (k & 7) | ((k & 16) >> 1) | ((k & 8) << 1);
}

// ---- async global->LDS tile copy (CDNA5 ASYNCcnt path) --------------------
#if defined(__gfx1250__) && __has_builtin(__builtin_amdgcn_global_load_async_to_lds_b128)
#define HAS_ASYNC_LDS 1
typedef int v4i __attribute__((ext_vector_type(4)));
typedef __attribute__((address_space(1))) v4i gv4i;   // global source
typedef __attribute__((address_space(3))) v4i lv4i;   // LDS destination
#define ASYNC_CP16(ldst, gsrc)                                          \
  __builtin_amdgcn_global_load_async_to_lds_b128(                       \
      (gv4i*)(gsrc), (lv4i*)(ldst), 0, 0)
template <int N> __device__ __forceinline__ void async_wait() {
#if __has_builtin(__builtin_amdgcn_s_wait_asynccnt)
  __builtin_amdgcn_s_wait_asynccnt(N);
#else
  asm volatile("s_wait_asynccnt %0" ::"i"(N) : "memory");
#endif
}
#else
#define HAS_ASYNC_LDS 0
#define ASYNC_CP16(ldst, gsrc) (*(uint4*)(ldst) = *(const uint4*)(gsrc))
template <int N> __device__ __forceinline__ void async_wait() {}
#endif

// ---- W_eff build ----------------------------------------------------------
__global__ __launch_bounds__(256) void weff_copy(float* __restrict__ weff,
                                                 const float* __restrict__ W) {
  size_t i = (size_t)blockIdx.x * blockDim.x + threadIdx.x;
  ((float4*)weff)[i] = ((const float4*)W)[i];
}

__global__ __launch_bounds__(256) void weff_scatter(float* __restrict__ weff,
                                                    const float* __restrict__ sw,
                                                    const int* __restrict__ idx,
                                                    int nnz) {
  int i = blockIdx.x * blockDim.x + threadIdx.x;
  if (i < nnz) atomicAdd(&weff[idx[i]], sw[i]);  // flat idx == row-major [o][i]
}

// ---- one-time bf16 hi/lo splits ------------------------------------------
__global__ __launch_bounds__(256) void split_plain(const float* __restrict__ src,
                                                   __bf16* __restrict__ hi,
                                                   __bf16* __restrict__ lo) {
  size_t i4 = ((size_t)blockIdx.x * 256 + threadIdx.x) * 4;
  const float4 v = *(const float4*)&src[i4];
  const float a[4] = {v.x, v.y, v.z, v.w};
  v4bf h, l;
#pragma unroll
  for (int e = 0; e < 4; ++e) {
    const __bf16 hh = (__bf16)a[e];
    h[e] = hh;
    l[e] = (__bf16)(a[e] - (float)hh);
  }
  *(v4bf*)&hi[i4] = h;
  *(v4bf*)&lo[i4] = l;
}

__global__ __launch_bounds__(256) void split_permA(const float* __restrict__ src,
                                                   __bf16* __restrict__ hi,
                                                   __bf16* __restrict__ lo) {
  size_t i4 = ((size_t)blockIdx.x * 256 + threadIdx.x) * 4;
  const float4 v = *(const float4*)&src[i4];
  const float a[4] = {v.x, v.y, v.z, v.w};
  v4bf h, l;
#pragma unroll
  for (int e = 0; e < 4; ++e) {
    const __bf16 hh = (__bf16)a[e];
    h[e] = hh;
    l[e] = (__bf16)(a[e] - (float)hh);
  }
  const size_t o = i4;  // plain K order
  *(v4bf*)&hi[o] = h;
  *(v4bf*)&lo[o] = l;
}

// ---- fast GEMM: precomputed bf16 planes + double-buffered async fill ------
__global__ __launch_bounds__(256) void gemm_bf16pre(
    const __bf16* __restrict__ Xh, const __bf16* __restrict__ Xl,
    const __bf16* __restrict__ Wh, const __bf16* __restrict__ Wl,
    const float* __restrict__ bias, float* __restrict__ out) {
  __shared__ __bf16 AsH[2][BM][BKP];
  __shared__ __bf16 AsL[2][BM][BKP];
  __shared__ __bf16 BsH[2][BN][BKP];
  __shared__ __bf16 BsL[2][BN][BKP];   // 2 x 40,960 B = 80 KB total

  const int tid  = threadIdx.x;
  const int lane = tid & 31;
  const int wave = tid >> 5;
  const int wr   = wave >> 2;   // 0..1
  const int wc   = wave & 3;    // 0..3
  const int half = lane >> 4;
  const int l16  = lane & 15;

  const int mBase = blockIdx.y * BM;
  const int nBase = blockIdx.x * BN;

  // per-thread copy assignment: 512 b128 chunks per tile, 2 per thread
  const int crow0 = tid >> 2;
  const int cce   = (tid & 3) * 8;     // bf16 elem offset within row

  auto issue_slab = [&](int k0, int buf) {
#pragma unroll
    for (int it = 0; it < 2; ++it) {
      const int row = crow0 + it * 64;
      const size_t gA = (size_t)(mBase + row) * IN_F + k0 + cce;
      const size_t gB = (size_t)(nBase + row) * IN_F + k0 + cce;
      ASYNC_CP16(&AsH[buf][row][cce], Xh + gA);
      ASYNC_CP16(&AsL[buf][row][cce], Xl + gA);
      ASYNC_CP16(&BsH[buf][row][cce], Wh + gB);
      ASYNC_CP16(&BsL[buf][row][cce], Wl + gB);
    }
  };

  v8f acc[4][2];
#pragma unroll
  for (int i = 0; i < 4; ++i)
#pragma unroll
    for (int j = 0; j < 2; ++j) acc[i][j] = (v8f){0.f,0.f,0.f,0.f,0.f,0.f,0.f,0.f};

  issue_slab(0, 0);   // prime the pipeline (8 async ops in flight)

  for (int s = 0; s < NSLAB; ++s) {
    const int buf = s & 1;
    if (s + 1 < NSLAB) {
      issue_slab((s + 1) * BK, buf ^ 1);  // other buffer: WAR-safe, barrier at
      async_wait<8>();                    // end of prev iter guards it.
    } else {                              // in-order completion: <=8 leaves
      async_wait<0>();                    // only slab s+1's copies in flight
    }
    __syncthreads();

    v16bf aH[4], aL[4], bH[2], bL[2];
#pragma unroll
    for (int i = 0; i < 4; ++i) {
      const int r = wr * 64 + i * 16 + l16;
      aH[i] = *(const v16bf*)&AsH[buf][r][half * 16];
      aL[i] = *(const v16bf*)&AsL[buf][r][half * 16];
    }
#pragma unroll
    for (int j = 0; j < 2; ++j) {
      const int c = wc * 32 + j * 16 + l16;
      bH[j] = *(const v16bf*)&BsH[buf][c][half * 16];
      bL[j] = *(const v16bf*)&BsL[buf][c][half * 16];
    }

#pragma unroll
    for (int i = 0; i < 4; ++i)
#pragma unroll
      for (int j = 0; j < 2; ++j) {
        acc[i][j] = __builtin_amdgcn_wmma_f32_16x16x32_bf16(
            false, aH[i], false, bH[j], (short)0, acc[i][j], false, false);
        acc[i][j] = __builtin_amdgcn_wmma_f32_16x16x32_bf16(
            false, aH[i], false, bL[j], (short)0, acc[i][j], false, false);
        acc[i][j] = __builtin_amdgcn_wmma_f32_16x16x32_bf16(
            false, aL[i], false, bH[j], (short)0, acc[i][j], false, false);
      }
    __syncthreads();  // all waves done reading buf before it is refilled
  }

#pragma unroll
  for (int j = 0; j < 2; ++j) {
    const int col = nBase + wc * 32 + j * 16 + l16;
    const float bv = bias[col];
#pragma unroll
    for (int i = 0; i < 4; ++i) {
      const int rb = mBase + wr * 64 + i * 16 + half * 8;
#pragma unroll
      for (int v = 0; v < 8; ++v)
        out[(size_t)(rb + v) * OUT_F + col] = acc[i][j][v] + bv;
    }
  }
}

// ---- fallback GEMM: on-the-fly split (needs only 64 MB ws) ----------------
__global__ __launch_bounds__(256) void gemm_bf16fly(
    const float* __restrict__ x, const float* __restrict__ weff,
    const float* __restrict__ bias, float* __restrict__ out) {
  __shared__ __bf16 AsH[BM][BKP];
  __shared__ __bf16 AsL[BM][BKP];
  __shared__ __bf16 BsH[BN][BKP];
  __shared__ __bf16 BsL[BN][BKP];

  const int tid  = threadIdx.x;
  const int lane = tid & 31;
  const int wave = tid >> 5;
  const int wr   = wave >> 2;
  const int wc   = wave & 3;
  const int half = lane >> 4;
  const int l16  = lane & 15;

  const int mBase = blockIdx.y * BM;
  const int nBase = blockIdx.x * BN;

  v8f acc[4][2];
#pragma unroll
  for (int i = 0; i < 4; ++i)
#pragma unroll
    for (int j = 0; j < 2; ++j) acc[i][j] = (v8f){0.f,0.f,0.f,0.f,0.f,0.f,0.f,0.f};

  for (int k0 = 0; k0 < IN_F; k0 += BK) {
    __syncthreads();
#pragma unroll
    for (int it = 0; it < 4; ++it) {
      const int t   = tid + it * 256;
      const int row = t >> 3;
      const int c4  = (t & 7) << 2;
      const float4 av = *(const float4*)&x[(size_t)(mBase + row) * IN_F + k0 + c4];
      const float4 bv = *(const float4*)&weff[(size_t)(nBase + row) * IN_F + k0 + c4];
      const float aa[4] = {av.x, av.y, av.z, av.w};
      const float bb[4] = {bv.x, bv.y, bv.z, bv.w};
      v4bf ah, al, bh, bl;
#pragma unroll
      for (int e = 0; e < 4; ++e) {
        const __bf16 h1 = (__bf16)aa[e];
        ah[e] = h1; al[e] = (__bf16)(aa[e] - (float)h1);
        const __bf16 h2 = (__bf16)bb[e];
        bh[e] = h2; bl[e] = (__bf16)(bb[e] - (float)h2);
      }
      const int pa = c4;
      *(v4bf*)&AsH[row][pa] = ah;
      *(v4bf*)&AsL[row][pa] = al;
      *(v4bf*)&BsH[row][c4] = bh;
      *(v4bf*)&BsL[row][c4] = bl;
    }
    __syncthreads();

    v16bf aH[4], aL[4], bH[2], bL[2];
#pragma unroll
    for (int i = 0; i < 4; ++i) {
      const int r = wr * 64 + i * 16 + l16;
      aH[i] = *(const v16bf*)&AsH[r][half * 16];
      aL[i] = *(const v16bf*)&AsL[r][half * 16];
    }
#pragma unroll
    for (int j = 0; j < 2; ++j) {
      const int c = wc * 32 + j * 16 + l16;
      bH[j] = *(const v16bf*)&BsH[c][half * 16];
      bL[j] = *(const v16bf*)&BsL[c][half * 16];
    }
#pragma unroll
    for (int i = 0; i < 4; ++i)
#pragma unroll
      for (int j = 0; j < 2; ++j) {
        acc[i][j] = __builtin_amdgcn_wmma_f32_16x16x32_bf16(
            false, aH[i], false, bH[j], (short)0, acc[i][j], false, false);
        acc[i][j] = __builtin_amdgcn_wmma_f32_16x16x32_bf16(
            false, aH[i], false, bL[j], (short)0, acc[i][j], false, false);
        acc[i][j] = __builtin_amdgcn_wmma_f32_16x16x32_bf16(
            false, aL[i], false, bH[j], (short)0, acc[i][j], false, false);
      }
  }

#pragma unroll
  for (int j = 0; j < 2; ++j) {
    const int col = nBase + wc * 32 + j * 16 + l16;
    const float bv = bias[col];
#pragma unroll
    for (int i = 0; i < 4; ++i) {
      const int rb = mBase + wr * 64 + i * 16 + half * 8;
#pragma unroll
      for (int v = 0; v < 8; ++v)
        out[(size_t)(rb + v) * OUT_F + col] = acc[i][j][v] + bv;
    }
  }
}

extern "C" void kernel_launch(void* const* d_in, const int* in_sizes, int n_in,
                              void* d_out, int out_size, void* d_ws, size_t ws_size,
                              hipStream_t stream) {
  const float* x    = (const float*)d_in[0];
  const float* W    = (const float*)d_in[1];
  const float* bias = (const float*)d_in[2];
  const float* sw   = (const float*)d_in[3];
  const int*   idx  = (const int*)d_in[4];
  const int    nnz  = in_sizes[3];

  float* out = (float*)d_out;
  char*  ws  = (char*)d_ws;

  const size_t WEFF_B = (size_t)OUT_F * IN_F * 4;   //  64 MB
  const size_t WH_B   = (size_t)OUT_F * IN_F * 2;   //  32 MB each
  const size_t XH_B   = (size_t)M_TOK * IN_F * 2;   // 128 MB each
  const size_t NEED   = WEFF_B + 2 * WH_B + 2 * XH_B;  // 384 MB

  float* weff = (float*)ws;
  weff_copy<<<(OUT_F * (size_t)IN_F / 4) / 256, 256, 0, stream>>>(weff, W);
  weff_scatter<<<(nnz + 255) / 256, 256, 0, stream>>>(weff, sw, idx, nnz);

  dim3 grid(OUT_F / BN, M_TOK / BM);   // (32, 128)

  if (ws_size >= NEED) {
    __bf16* Wh = (__bf16*)(ws + WEFF_B);
    __bf16* Wl = (__bf16*)(ws + WEFF_B + WH_B);
    __bf16* Xh = (__bf16*)(ws + WEFF_B + 2 * WH_B);
    __bf16* Xl = (__bf16*)(ws + WEFF_B + 2 * WH_B + XH_B);
    split_plain<<<(OUT_F * (size_t)IN_F / 4) / 256, 256, 0, stream>>>(weff, Wh, Wl);
    split_permA<<<((size_t)M_TOK * IN_F / 4) / 256, 256, 0, stream>>>(x, Xh, Xl);
    gemm_bf16pre<<<grid, 256, 0, stream>>>(Xh, Xl, Wh, Wl, bias, out);
  } else {
    gemm_bf16fly<<<grid, 256, 0, stream>>>(x, weff, bias, out);
  }
}
